// WindowAttention_738734375474
// MI455X (gfx1250) — matrix operand order repaired and checked by reading, not compiled, not run on hardware
//
#include <hip/hip_runtime.h>
#include <hip/hip_bf16.h>

typedef _Float16 f16;
typedef __attribute__((ext_vector_type(8)))  _Float16 v8h;
typedef __attribute__((ext_vector_type(16))) _Float16 v16h;
typedef __attribute__((ext_vector_type(8)))  float    v8f;

#define S49   49
#define SPAD  64
#define DIM   256
#define NH    8
#define HD    32
#define NWIN  64
#define SCALE 0.17677669529663689f

// LDS layout (bytes). Strides in f16 elements, all keep 16B alignment for b128
// and give conflict-free 64-bank access for 16-lane b128 loads.
#define X_STRIDE  264   // [64][264] f16 : x staging, later attn_out
#define QK_STRIDE 40    // per head [64][40] f16
#define VT_STRIDE 72    // per head [32][72] f16 (V transposed: [dim][token])
#define P_STRIDE  72    // per head [64][72] f16

#define X_OFF 0
#define Q_OFF 33792
#define K_OFF (33792 + 40960)
#define V_OFF (33792 + 81920)
#define P_OFF 33792                       // overlays Q+K after regs snapshot
#define SMEM_BYTES (33792 + 81920 + 36864) // 152576 <= 320KB WGP LDS

__device__ __forceinline__ v16h cat16(v8h lo, v8h hi) {
  return __builtin_shufflevector(lo, hi, 0,1,2,3,4,5,6,7,8,9,10,11,12,13,14,15);
}
// A-matrix 16x32 f16 tile: row-major [M][K] source. lane&15 = M, half picks K set.
__device__ __forceinline__ v16h load_a(const f16* base, int stride, int m0, int k0,
                                       int lr, int hf) {
  const f16* p = base + (m0 + lr) * stride + k0 + hf * 8;
  return cat16(*(const v8h*)p, *(const v8h*)(p + 16));
}
// B-matrix 32x16 f16 tile from B^T row-major [N][K]. lane&15 = N, half picks K set.
__device__ __forceinline__ v16h load_bt(const f16* baseT, int stride, int n0, int k0,
                                        int lr, int hf) {
  const f16* p = baseT + (n0 + lr) * stride + k0 + hf * 8;
  return cat16(*(const v8h*)p, *(const v8h*)(p + 16));
}
__device__ __forceinline__ v8f wmma16(v16h a, v16h b, v8f c) {
  return __builtin_amdgcn_wmma_f32_16x16x32_f16(false, a, false, b, (short)0, c,
                                                false, false);
}

// ---- prep: transpose weights to f16 W^T [N][K] in workspace (L2 resident) ----
__global__ __launch_bounds__(256) void prep_weights(
    const float* __restrict__ qkv_w, const float* __restrict__ proj_w,
    f16* __restrict__ wqkvT, f16* __restrict__ wprojT) {
  int idx = blockIdx.x * 256 + threadIdx.x;
  if (idx < 768 * 256) {
    int n = idx >> 8, k = idx & 255;
    wqkvT[idx] = (f16)qkv_w[k * 768 + n];
  } else {
    int j = idx - 768 * 256;
    if (j < 256 * 256) {
      int n = j >> 8, k = j & 255;
      wprojT[j] = (f16)proj_w[k * 256 + n];
    }
  }
}

// ---- fused window attention: one block = one window (49 tokens, 256 ch) ----
__global__ __launch_bounds__(256) void fused_window_attn(
    const float* __restrict__ x, const float* __restrict__ maskg,
    const float* __restrict__ qkv_b, const float* __restrict__ proj_b,
    const float* __restrict__ bias_table, const int* __restrict__ rel_idx,
    const f16* __restrict__ wqkvT, const f16* __restrict__ wprojT,
    float* __restrict__ out) {
  __shared__ __align__(16) unsigned char smem[SMEM_BYTES];
  f16* xl  = (f16*)(smem + X_OFF);
  f16* ql  = (f16*)(smem + Q_OFF);
  f16* kl  = (f16*)(smem + K_OFF);
  f16* vtl = (f16*)(smem + V_OFF);
  f16* pl  = (f16*)(smem + P_OFF);

  const int tid  = threadIdx.x;
  const int wave = tid >> 5;
  const int lane = tid & 31;
  const int hf   = lane >> 4;
  const int lr   = lane & 15;
  const int blk  = blockIdx.x;
  const int widx = blk & (NWIN - 1);

  // ---- stage x -> f16 LDS, zero pad rows 49..63 ----
  const float* xg = x + (size_t)blk * (S49 * DIM);
  for (int i = tid; i < S49 * DIM; i += 256) {
    int r = i >> 8, c = i & 255;
    xl[r * X_STRIDE + c] = (f16)xg[i];
  }
  for (int i = tid; i < (SPAD - S49) * DIM; i += 256) {
    int r = S49 + (i >> 8), c = i & 255;
    xl[r * X_STRIDE + c] = (f16)0.f;
  }
  __syncthreads();

  // ---- QKV GEMM: [64,256] x [256,768]; wave w owns N-tiles 6w..6w+5 ----
  for (int i = 0; i < 6; ++i) {
    const int n0 = (wave * 6 + i) * 16;
    v8f a0 = {}, a1 = {}, a2 = {}, a3 = {};
#pragma unroll
    for (int ks = 0; ks < 8; ++ks) {
      const int k0 = ks * 32;
      v16h bt = load_bt(wqkvT, DIM, n0, k0, lr, hf);   // global (L2) b128 x2
      a0 = wmma16(load_a(xl, X_STRIDE, 0,  k0, lr, hf), bt, a0);
      a1 = wmma16(load_a(xl, X_STRIDE, 16, k0, lr, hf), bt, a1);
      a2 = wmma16(load_a(xl, X_STRIDE, 32, k0, lr, hf), bt, a2);
      a3 = wmma16(load_a(xl, X_STRIDE, 48, k0, lr, hf), bt, a3);
    }
    const int c   = n0 + lr;         // tile spans one segment/head: wave-uniform branch
    const float bc = qkv_b[c];
    v8f accs[4] = {a0, a1, a2, a3};
#pragma unroll
    for (int mt = 0; mt < 4; ++mt) {
#pragma unroll
      for (int r = 0; r < 8; ++r) {
        const int tok = mt * 16 + r + hf * 8;
        const float v = accs[mt][r] + bc;
        if (c < 256) {
          const int h = c >> 5, d = c & 31;
          ql[h * (SPAD * QK_STRIDE) + tok * QK_STRIDE + d] = (f16)(v * SCALE);
        } else if (c < 512) {
          const int h = (c - 256) >> 5, d = c & 31;
          kl[h * (SPAD * QK_STRIDE) + tok * QK_STRIDE + d] = (f16)v;
        } else {
          const int h = (c - 512) >> 5, d = c & 31;
          vtl[h * (HD * VT_STRIDE) + d * VT_STRIDE + tok] = (f16)v;   // transposed
        }
      }
    }
  }
  __syncthreads();

  // ---- attention: wave w owns head w ----
  const int h = wave;
  const f16* qh = ql  + h * (SPAD * QK_STRIDE);
  const f16* kh = kl  + h * (SPAD * QK_STRIDE);
  const f16* vh = vtl + h * (HD * VT_STRIDE);
  f16*       ph = pl  + h * (SPAD * P_STRIDE);

  // snapshot Q/K tiles into registers (P will overlay their LDS region)
  v16h qts[4], kts[4];
#pragma unroll
  for (int t = 0; t < 4; ++t) {
    qts[t] = load_a (qh, QK_STRIDE, t * 16, 0, lr, hf);
    kts[t] = load_bt(kh, QK_STRIDE, t * 16, 0, lr, hf);
  }
  __syncthreads();   // all waves hold Q/K; safe to overwrite with P

  const float* mp = maskg + (size_t)widx * (S49 * S49);
  const v8f zero = {};
#pragma unroll
  for (int mt = 0; mt < 4; ++mt) {
    v8f s[4];
    s[0] = wmma16(qts[mt], kts[0], zero);
    s[1] = wmma16(qts[mt], kts[1], zero);
    s[2] = wmma16(qts[mt], kts[2], zero);
    s[3] = wmma16(qts[mt], kts[3], zero);
#pragma unroll
    for (int r = 0; r < 8; ++r) {
      const int q = mt * 16 + r + hf * 8;
#pragma unroll
      for (int nt = 0; nt < 4; ++nt) {
        const int n = nt * 16 + lr;
        float val = s[nt][r];
        if (q >= S49)      val = 0.f;      // dead row: keep finite
        else if (n >= S49) val = -1e30f;   // masked pad key
        else val += bias_table[rel_idx[q * S49 + n] * NH + h] + mp[q * S49 + n];
        s[nt][r] = val;
      }
      // softmax over the 64 columns: 4 tiles x 16 lanes (within half)
      float m = fmaxf(fmaxf(s[0][r], s[1][r]), fmaxf(s[2][r], s[3][r]));
      m = fmaxf(m, __shfl_xor(m, 1));
      m = fmaxf(m, __shfl_xor(m, 2));
      m = fmaxf(m, __shfl_xor(m, 4));
      m = fmaxf(m, __shfl_xor(m, 8));
      float sum = 0.f;
#pragma unroll
      for (int nt = 0; nt < 4; ++nt) {
        const float e = __expf(s[nt][r] - m);
        s[nt][r] = e;
        sum += e;
      }
      sum += __shfl_xor(sum, 1);
      sum += __shfl_xor(sum, 2);
      sum += __shfl_xor(sum, 4);
      sum += __shfl_xor(sum, 8);
      const float inv = 1.f / sum;
#pragma unroll
      for (int nt = 0; nt < 4; ++nt)
        ph[q * P_STRIDE + nt * 16 + lr] = (f16)(s[nt][r] * inv);
    }
  }

  // ---- O = P[64,64] @ V[64,32] ; write f16 into x region (attn_out) ----
#pragma unroll
  for (int mt = 0; mt < 4; ++mt) {
#pragma unroll
    for (int nt2 = 0; nt2 < 2; ++nt2) {
      v8f o = {};
#pragma unroll
      for (int ks = 0; ks < 2; ++ks) {
        v16h a = load_a (ph, P_STRIDE,  mt * 16,  ks * 32, lr, hf);
        v16h b = load_bt(vh, VT_STRIDE, nt2 * 16, ks * 32, lr, hf);
        o = wmma16(a, b, o);
      }
#pragma unroll
      for (int r = 0; r < 8; ++r) {
        const int tok = mt * 16 + r + hf * 8;
        xl[tok * X_STRIDE + h * HD + nt2 * 16 + lr] = (f16)o[r];
      }
    }
  }
  __syncthreads();

  // ---- proj GEMM: [64,256] x [256,256]; wave w owns N-tiles 2w, 2w+1 ----
  float* og = out + (size_t)blk * (S49 * DIM);
#pragma unroll
  for (int i = 0; i < 2; ++i) {
    const int n0 = (wave * 2 + i) * 16;
    v8f acc[4] = {};
#pragma unroll
    for (int ks = 0; ks < 8; ++ks) {
      const int k0 = ks * 32;
      v16h b = load_bt(wprojT, DIM, n0, k0, lr, hf);
#pragma unroll
      for (int mt = 0; mt < 4; ++mt)
        acc[mt] = wmma16(load_a(xl, X_STRIDE, mt * 16, k0, lr, hf), b, acc[mt]);
    }
    const float bc = proj_b[n0 + lr];
#pragma unroll
    for (int mt = 0; mt < 4; ++mt) {
#pragma unroll
      for (int r = 0; r < 8; ++r) {
        const int q = mt * 16 + r + hf * 8;
        if (q < S49) og[q * DIM + n0 + lr] = acc[mt][r] + bc;
      }
    }
  }
}

extern "C" void kernel_launch(void* const* d_in, const int* in_sizes, int n_in,
                              void* d_out, int out_size, void* d_ws, size_t ws_size,
                              hipStream_t stream) {
  const float* x          = (const float*)d_in[0];
  const float* mask       = (const float*)d_in[1];
  const float* qkv_w      = (const float*)d_in[2];
  const float* qkv_b      = (const float*)d_in[3];
  const float* proj_w     = (const float*)d_in[4];
  const float* proj_b     = (const float*)d_in[5];
  const float* bias_table = (const float*)d_in[6];
  const int*   rel_idx    = (const int*)d_in[7];

  f16* wqkvT  = (f16*)d_ws;              // [768][256] f16 = 384 KB
  f16* wprojT = wqkvT + 768 * 256;       // [256][256] f16 = 128 KB

  prep_weights<<<(768 * 256 + 256 * 256 + 255) / 256, 256, 0, stream>>>(
      qkv_w, proj_w, wqkvT, wprojT);
  fused_window_attn<<<4096, 256, 0, stream>>>(
      x, mask, qkv_b, proj_b, bias_table, rel_idx, wqkvT, wprojT, (float*)d_out);
}
